// MixtureOfExperts_16466904613586
// MI455X (gfx1250) — matrix operand order repaired and checked by reading, not compiled, not run on hardware
//
#include <hip/hip_runtime.h>
#include <hip/hip_bf16.h>

// ---------------- problem constants ----------------
#define NTOK 4096   // B*S
#define DDIM 1024
#define HDIM 2048
#define ODIM 1024
#define NEXP 8

typedef __attribute__((ext_vector_type(16))) __bf16 v16bf;
typedef __attribute__((ext_vector_type(8)))  float  v8f;

#define LDS_STRIDE 40   // bf16 elems per tile row (80B, padded from 32; 16B aligned)

// ---- WMMA helper: D = A(16x32 bf16) * B(32x16 bf16) + C(16x16 f32) ----
__device__ __forceinline__ v8f wmma_bf16(v16bf a, v16bf b, v8f c) {
    return __builtin_amdgcn_wmma_f32_16x16x32_bf16(
        /*neg_a=*/false, a, /*neg_b=*/false, b,
        /*c_mod=*/(short)0, c, /*reuse_a=*/false, /*reuse_b=*/false);
}

// fast SiLU: g * sigmoid(g) using hardware v_rcp_f32 (~1 ulp, plenty for bf16 out)
__device__ __forceinline__ float silu_fast(float g) {
    return g * __builtin_amdgcn_rcpf(1.f + __expf(-g));
}

// A fragment (16x32, MxK). Lane L: m = L%16, half = L/16.
// VGPR i in 0..3 -> k = half*8 + 2i..2i+1 ; i in 4..7 -> k = 16 + half*8 + ...
// => two contiguous 8-bf16 runs: bytes [half*16, +15] and [32+half*16, +15].
__device__ __forceinline__ v16bf load_frag_A(const __bf16* tile, int lane) {
    int m = lane & 15, half = lane >> 4;
    const __bf16* row = tile + m * LDS_STRIDE;
    union { v16bf v; uint4 u[2]; } f;
    f.u[0] = *(const uint4*)(row + half * 8);
    f.u[1] = *(const uint4*)(row + 16 + half * 8);
    return f.v;
}

// B fragment (32x16, KxN), stored in LDS as [n][k]. Lane L: n = L%16, half = L/16.
// Same k order as A: k = half*8 .. +7, then 16 + half*8 .. +7  => two 16-byte runs.
__device__ __forceinline__ v16bf load_frag_B(const __bf16* tile, int lane) {
    int n = lane & 15, half = lane >> 4;
    const __bf16* row = tile + n * LDS_STRIDE;
    union { v16bf v; uint4 u[2]; } f;
    f.u[0] = *(const uint4*)(row + half * 8);
    f.u[1] = *(const uint4*)(row + 16 + half * 8);
    return f.v;
}

// ---------------- fp32 -> bf16 conversion ----------------
__global__ void moe_cvt_bf16(const float* __restrict__ src,
                             __bf16* __restrict__ dst, int n) {
    int i = blockIdx.x * blockDim.x + threadIdx.x;
    int stride = gridDim.x * blockDim.x;
    for (; i < n; i += stride) dst[i] = (__bf16)src[i];
}

// ---------------- gating: logits, softmax, top-2, renormalize ----------------
__global__ __launch_bounds__(256) void moe_gate(
    const float* __restrict__ x,      // [NTOK, DDIM]
    const float* __restrict__ Wr,     // [NEXP, DDIM]
    const float* __restrict__ br,     // [NEXP]
    float* __restrict__ logits_out,   // [NTOK, NEXP]
    int*   __restrict__ topk_out,     // [NTOK, 2]
    float* __restrict__ wgate)        // [NTOK, NEXP] sparse renormalized
{
    const int t = blockIdx.x;
    const int e = threadIdx.x >> 5;   // 8 waves, one expert each
    const int lane = threadIdx.x & 31;
    __shared__ float lg[NEXP];

    float s = 0.f;
    const float* xr = x + (size_t)t * DDIM;
    const float* wr = Wr + (size_t)e * DDIM;
    for (int d = lane; d < DDIM; d += 32) s += xr[d] * wr[d];
    for (int off = 16; off > 0; off >>= 1) s += __shfl_down(s, off, 32);
    if (lane == 0) lg[e] = s + br[e];
    __syncthreads();

    if (threadIdx.x == 0) {
        float mx = lg[0];
        for (int i = 1; i < NEXP; ++i) mx = fmaxf(mx, lg[i]);
        float sw[NEXP]; float sum = 0.f;
        for (int i = 0; i < NEXP; ++i) { sw[i] = __expf(lg[i] - mx); sum += sw[i]; }
        float inv = 1.f / sum;
        for (int i = 0; i < NEXP; ++i) sw[i] *= inv;
        int i0 = 0;
        for (int i = 1; i < NEXP; ++i) if (sw[i] > sw[i0]) i0 = i;
        int i1 = (i0 == 0) ? 1 : 0;
        for (int i = 0; i < NEXP; ++i) if (i != i0 && sw[i] > sw[i1]) i1 = i;
        float norm = 1.f / (sw[i0] + sw[i1]);
        for (int i = 0; i < NEXP; ++i) wgate[(size_t)t * NEXP + i] = 0.f;
        wgate[(size_t)t * NEXP + i0] = sw[i0] * norm;
        wgate[(size_t)t * NEXP + i1] = sw[i1] * norm;
        for (int i = 0; i < NEXP; ++i) logits_out[(size_t)t * NEXP + i] = lg[i];
        topk_out[t * 2 + 0] = i0;
        topk_out[t * 2 + 1] = i1;
    }
}

// ---------------- GEMM1: U = silu(X*Wg^T) * (X*Wv^T) * w[t,e]  (bf16 out) ----
// Block tile 128(tokens) x 64(H). 8 waves: 4 along M x 2 along N, 32x32 each.
__global__ __launch_bounds__(256) void moe_gemm1(
    const __bf16* __restrict__ Xb,   // [NTOK, DDIM]
    const __bf16* __restrict__ Wg,   // [HDIM, DDIM] (this expert)
    const __bf16* __restrict__ Wv,   // [HDIM, DDIM]
    const float*  __restrict__ wgate,// [NTOK, NEXP]
    __bf16* __restrict__ U,          // [NTOK, HDIM]
    int expert)                      // 0..7, or 8 = shared (weight 1)
{
    __shared__ __align__(16) __bf16 As[128 * LDS_STRIDE];
    __shared__ __align__(16) __bf16 Bg[64 * LDS_STRIDE];
    __shared__ __align__(16) __bf16 Bv[64 * LDS_STRIDE];
    __shared__ float wT[128];

    const int tid = threadIdx.x;
    const int lane = tid & 31, wave = tid >> 5;
    const int wM = wave & 3, wN = wave >> 2;
    const int t0 = blockIdx.x * 128;
    const int h0 = blockIdx.y * 64;

    if (tid < 128)
        wT[tid] = (expert < NEXP) ? wgate[(size_t)(t0 + tid) * NEXP + expert] : 1.0f;

    v8f accG[2][2], accV[2][2];
    const v8f vz = {0.f, 0.f, 0.f, 0.f, 0.f, 0.f, 0.f, 0.f};
    for (int mi = 0; mi < 2; ++mi)
        for (int ni = 0; ni < 2; ++ni) { accG[mi][ni] = vz; accV[mi][ni] = vz; }

    for (int kk = 0; kk < DDIM; kk += 32) {
        __syncthreads();
        // A tile: 128x32 bf16 = 512 chunks of 8; 2 per thread
        for (int c = 0; c < 2; ++c) {
            int chunk = tid + c * 256;
            int row = chunk >> 2, col = (chunk & 3) * 8;
            *(uint4*)&As[row * LDS_STRIDE + col] =
                *(const uint4*)&Xb[(size_t)(t0 + row) * DDIM + kk + col];
        }
        // B tiles: 64x32 each = 256 chunks; 1 per thread per tile
        {
            int row = tid >> 2, col = (tid & 3) * 8;
            *(uint4*)&Bg[row * LDS_STRIDE + col] =
                *(const uint4*)&Wg[(size_t)(h0 + row) * DDIM + kk + col];
            *(uint4*)&Bv[row * LDS_STRIDE + col] =
                *(const uint4*)&Wv[(size_t)(h0 + row) * DDIM + kk + col];
        }
        __syncthreads();

        v16bf a[2], bg[2], bv[2];
        for (int mi = 0; mi < 2; ++mi)
            a[mi] = load_frag_A(&As[(wM * 32 + mi * 16) * LDS_STRIDE], lane);
        for (int ni = 0; ni < 2; ++ni) {
            bg[ni] = load_frag_B(&Bg[(wN * 32 + ni * 16) * LDS_STRIDE], lane);
            bv[ni] = load_frag_B(&Bv[(wN * 32 + ni * 16) * LDS_STRIDE], lane);
        }
        for (int mi = 0; mi < 2; ++mi)
            for (int ni = 0; ni < 2; ++ni) {
                accG[mi][ni] = wmma_bf16(a[mi], bg[ni], accG[mi][ni]);
                accV[mi][ni] = wmma_bf16(a[mi], bv[ni], accV[mi][ni]);
            }
    }
    __syncthreads();

    // epilogue: u = silu(g)*v * w[t,e]
    const int n = lane & 15, halfsel = lane >> 4;
    for (int mi = 0; mi < 2; ++mi)
        for (int ni = 0; ni < 2; ++ni)
            for (int i = 0; i < 8; ++i) {
                int m = wM * 32 + mi * 16 + i + halfsel * 8;
                float g = accG[mi][ni][i];
                float v = accV[mi][ni][i];
                float u = silu_fast(g) * v * wT[m];
                int h = h0 + wN * 32 + ni * 16 + n;
                U[(size_t)(t0 + m) * HDIM + h] = (__bf16)u;
            }
}

// ---------------- GEMM2: out (+)= U * Wo^T  (+ combined bias on first) -------
__global__ __launch_bounds__(256) void moe_gemm2(
    const __bf16* __restrict__ U,    // [NTOK, HDIM]
    const __bf16* __restrict__ Wo,   // [ODIM, HDIM] (this expert)
    const float*  __restrict__ wgate,// [NTOK, NEXP]
    const float*  __restrict__ bo,   // [NEXP, ODIM]
    const float*  __restrict__ bos,  // [ODIM]
    float* __restrict__ out,         // [NTOK, ODIM]
    int first)
{
    __shared__ __align__(16) __bf16 As[128 * LDS_STRIDE];
    __shared__ __align__(16) __bf16 Bs[64 * LDS_STRIDE];
    __shared__ float wT[128 * NEXP];
    __shared__ float boT[NEXP * 64];
    __shared__ float bosT[64];

    const int tid = threadIdx.x;
    const int lane = tid & 31, wave = tid >> 5;
    const int wM = wave & 3, wN = wave >> 2;
    const int t0 = blockIdx.x * 128;
    const int o0 = blockIdx.y * 64;

    // bias / gate tiles (only consumed when first != 0, cheap either way)
    for (int c = 0; c < 4; ++c) {
        int idx = tid + c * 256;           // 0..1023 == 128*8, contiguous [t][e]
        wT[idx] = wgate[(size_t)t0 * NEXP + idx];
    }
    if (tid < NEXP * 64) boT[tid] = bo[(size_t)(tid >> 6) * ODIM + o0 + (tid & 63)];
    if (tid < 64) bosT[tid] = bos[o0 + tid];

    v8f acc[2][2];
    const v8f vz = {0.f, 0.f, 0.f, 0.f, 0.f, 0.f, 0.f, 0.f};
    for (int mi = 0; mi < 2; ++mi)
        for (int ni = 0; ni < 2; ++ni) acc[mi][ni] = vz;

    for (int kk = 0; kk < HDIM; kk += 32) {
        __syncthreads();
        for (int c = 0; c < 2; ++c) {
            int chunk = tid + c * 256;
            int row = chunk >> 2, col = (chunk & 3) * 8;
            *(uint4*)&As[row * LDS_STRIDE + col] =
                *(const uint4*)&U[(size_t)(t0 + row) * HDIM + kk + col];
        }
        {
            int row = tid >> 2, col = (tid & 3) * 8;
            *(uint4*)&Bs[row * LDS_STRIDE + col] =
                *(const uint4*)&Wo[(size_t)(o0 + row) * HDIM + kk + col];
        }
        __syncthreads();

        v16bf a[2], b[2];
        for (int mi = 0; mi < 2; ++mi)
            a[mi] = load_frag_A(&As[(wM * 32 + mi * 16) * LDS_STRIDE], lane);
        for (int ni = 0; ni < 2; ++ni)
            b[ni] = load_frag_B(&Bs[(wN * 32 + ni * 16) * LDS_STRIDE], lane);
        for (int mi = 0; mi < 2; ++mi)
            for (int ni = 0; ni < 2; ++ni)
                acc[mi][ni] = wmma_bf16(a[mi], b[ni], acc[mi][ni]);
    }
    __syncthreads();

    const int n = lane & 15, halfsel = lane >> 4;
    for (int mi = 0; mi < 2; ++mi)
        for (int ni = 0; ni < 2; ++ni)
            for (int i = 0; i < 8; ++i) {
                int m = wM * 32 + mi * 16 + i + halfsel * 8;
                int oc = wN * 32 + ni * 16 + n;
                size_t oidx = (size_t)(t0 + m) * ODIM + o0 + oc;
                float val = acc[mi][ni][i];
                if (first) {
                    float bterm = bosT[oc];
                    for (int e = 0; e < NEXP; ++e)
                        bterm += wT[m * NEXP + e] * boT[e * 64 + oc];
                    out[oidx] = val + bterm;
                } else {
                    out[oidx] += val;
                }
            }
}

// ---------------- launcher ----------------
extern "C" void kernel_launch(void* const* d_in, const int* in_sizes, int n_in,
                              void* d_out, int out_size, void* d_ws, size_t ws_size,
                              hipStream_t stream) {
    (void)in_sizes; (void)n_in; (void)out_size; (void)ws_size;

    const float* x    = (const float*)d_in[0];
    const float* Wr   = (const float*)d_in[1];
    const float* br   = (const float*)d_in[2];
    const float* Wg   = (const float*)d_in[3];
    const float* Wv   = (const float*)d_in[4];
    const float* Wo   = (const float*)d_in[5];
    const float* bo   = (const float*)d_in[6];
    const float* Wg_s = (const float*)d_in[7];
    const float* Wv_s = (const float*)d_in[8];
    const float* Wo_s = (const float*)d_in[9];
    const float* bo_s = (const float*)d_in[10];

    // d_out tuple layout: output f32 [NTOK*ODIM], gate_logits f32 [NTOK*NEXP],
    // topk_i int32 [NTOK*2]
    float* out        = (float*)d_out;
    float* logits_out = out + (size_t)NTOK * ODIM;
    int*   topk_out   = (int*)(logits_out + (size_t)NTOK * NEXP);

    // workspace layout (bytes)
    char* ws = (char*)d_ws;
    __bf16* Xb   = (__bf16*)(ws + 0);                    //  8.0 MB
    __bf16* WgB  = (__bf16*)(ws + 8388608ull);           // 32.0 MB
    __bf16* WvB  = (__bf16*)(ws + 41943040ull);          // 32.0 MB
    __bf16* WoB  = (__bf16*)(ws + 75497472ull);          // 32.0 MB
    __bf16* WgSB = (__bf16*)(ws + 109051904ull);         //  4.0 MB
    __bf16* WvSB = (__bf16*)(ws + 113246208ull);         //  4.0 MB
    __bf16* WoSB = (__bf16*)(ws + 117440512ull);         //  4.0 MB
    float*  wgate= (float*) (ws + 121634816ull);         //  128 KB
    __bf16* Ubuf = (__bf16*)(ws + 121765888ull);         // 16.0 MB  (one expert)

    const int CT = 256, CB = 2048;
    moe_cvt_bf16<<<CB, CT, 0, stream>>>(x,    Xb,   NTOK * DDIM);
    moe_cvt_bf16<<<CB, CT, 0, stream>>>(Wg,   WgB,  NEXP * HDIM * DDIM);
    moe_cvt_bf16<<<CB, CT, 0, stream>>>(Wv,   WvB,  NEXP * HDIM * DDIM);
    moe_cvt_bf16<<<CB, CT, 0, stream>>>(Wo,   WoB,  NEXP * ODIM * HDIM);
    moe_cvt_bf16<<<CB, CT, 0, stream>>>(Wg_s, WgSB, HDIM * DDIM);
    moe_cvt_bf16<<<CB, CT, 0, stream>>>(Wv_s, WvSB, HDIM * DDIM);
    moe_cvt_bf16<<<CB, CT, 0, stream>>>(Wo_s, WoSB, ODIM * HDIM);

    moe_gate<<<NTOK, 256, 0, stream>>>(x, Wr, br, logits_out, topk_out, wgate);

    dim3 g1(NTOK / 128, HDIM / 64);
    dim3 g2(NTOK / 128, ODIM / 64);
    for (int e = 0; e < NEXP; ++e) {
        moe_gemm1<<<g1, 256, 0, stream>>>(
            Xb, WgB + (size_t)e * HDIM * DDIM, WvB + (size_t)e * HDIM * DDIM,
            wgate, Ubuf, e);
        moe_gemm2<<<g2, 256, 0, stream>>>(
            Ubuf, WoB + (size_t)e * ODIM * HDIM, wgate, bo, bo_s, out,
            /*first=*/(e == 0) ? 1 : 0);
    }
    // shared expert (gate weight 1, bias already applied on first pass)
    moe_gemm1<<<g1, 256, 0, stream>>>(Xb, WgSB, WvSB, wgate, Ubuf, NEXP);
    moe_gemm2<<<g2, 256, 0, stream>>>(Ubuf, WoSB, wgate, bo, bo_s, out, 0);
}
